// MultiHeadAttention_32573031973333
// MI455X (gfx1250) — matrix operand order repaired and checked by reading, not compiled, not run on hardware
//
#include <hip/hip_runtime.h>

// ---------------------------------------------------------------------------
// MultiHeadAttention for gfx1250 (MI455X): bf16 WMMA pipeline with a fused
// scores+softmax+AV kernel that keeps each 16-row weight strip in LDS.
// B=2, S=2048, E=1024, H=16, DK=DV=64.
// d_out = [ out (B*S*E f32) | weights (B*H*S*S f32) ]
// ---------------------------------------------------------------------------

typedef __bf16 bf16_t;
typedef bf16_t v16bf __attribute__((ext_vector_type(16)));
typedef float  v8f   __attribute__((ext_vector_type(8)));

constexpr int CB = 2, CS = 2048, CE = 1024, CH = 16, CD = 64;
constexpr int CN = CH * CD;                          // 1024 (= E)
constexpr size_t OUT_ELEMS = (size_t)CB * CS * CE;   // 4,194,304

// Workspace layout (bf16 elements), ~48 MB total.
constexpr size_t OFF_XH = 0;                                   // x in bf16 (B*S, E)
constexpr size_t OFF_WQ = OFF_XH + (size_t)CB * CS * CE;       // w_q^T bf16 (N, E)
constexpr size_t OFF_WK = OFF_WQ + (size_t)CE * CN;
constexpr size_t OFF_WV = OFF_WK + (size_t)CE * CN;
constexpr size_t OFF_WO = OFF_WV + (size_t)CE * CN;            // w_o^T bf16 (E, N)
constexpr size_t OFF_Q  = OFF_WO + (size_t)CN * CE;            // Q bf16 (b,h,s,d)
constexpr size_t OFF_K  = OFF_Q  + (size_t)CB * CH * CS * CD;  // K bf16 (b,h,s,d)
constexpr size_t OFF_VT = OFF_K  + (size_t)CB * CH * CS * CD;  // V^T bf16 (b,h,d,s)
constexpr size_t OFF_AV = OFF_VT + (size_t)CB * CH * CS * CD;  // AV bf16 (b,s,h*d)

__device__ __forceinline__ int lane_id() { return (int)(threadIdx.x & 31); }

// A fragment: 16x32 bf16 (MxK), row-major source, stride lda.
// Lane l (m = l&15, kh = l>>4): K = {8*kh+0..7} in v[0..7], {16+8*kh+0..7} in v[8..15].
__device__ __forceinline__ v16bf load_a_bf16(const bf16_t* p, int lda) {
  const int l = lane_id();
  const bf16_t* row = p + (size_t)(l & 15) * lda + 8 * (l >> 4);
  v16bf a;
#pragma unroll
  for (int j = 0; j < 8; ++j) { a[j] = row[j]; a[j + 8] = row[16 + j]; }
  return a;
}

// B fragment: 32x16 bf16 (KxN) where src holds B^T row-major (N rows, stride lds):
// B[k][n] = src[n*lds + k]. Lane l (n = l&15, h = l>>4): K = {8*h+0..7} in v[0..7], {16+8*h+0..7} in v[8..15].
__device__ __forceinline__ v16bf load_bt_bf16(const bf16_t* p, int lds) {
  const int l = lane_id();
  const bf16_t* row = p + (size_t)(l & 15) * lds + 8 * (l >> 4);
  v16bf b;
#pragma unroll
  for (int j = 0; j < 8; ++j) { b[j] = row[j]; b[j + 8] = row[16 + j]; }
  return b;
}

__device__ __forceinline__ v8f wmma_bf16(v16bf a, v16bf b, v8f c) {
  return __builtin_amdgcn_wmma_f32_16x16x32_bf16(false, a, false, b, (short)0, c,
                                                 false, false);
}

// ---------------------------------------------------------------------------
// Kernel 1: x f32 -> bf16.
__global__ void cvt_x_kernel(const float* __restrict__ x, bf16_t* __restrict__ xh,
                             int n) {
  int i = blockIdx.x * blockDim.x + threadIdx.x;
  if (i < n) xh[i] = (bf16_t)x[i];
}

// Kernel 2: transpose-convert the four 1024x1024 f32 weight matrices to bf16.
__global__ void cvt_wT_kernel(const float* __restrict__ wq,
                              const float* __restrict__ wk,
                              const float* __restrict__ wv,
                              const float* __restrict__ wo,
                              bf16_t* __restrict__ ws) {
  __shared__ float tile[32][33];
  const float* src;
  bf16_t* dst;
  switch (blockIdx.z) {
    case 0:  src = wq; dst = ws + OFF_WQ; break;
    case 1:  src = wk; dst = ws + OFF_WK; break;
    case 2:  src = wv; dst = ws + OFF_WV; break;
    default: src = wo; dst = ws + OFF_WO; break;
  }
  const int c0 = blockIdx.x * 32, r0 = blockIdx.y * 32;
  for (int i = threadIdx.y; i < 32; i += blockDim.y)
    tile[i][threadIdx.x] = src[(size_t)(r0 + i) * 1024 + c0 + threadIdx.x];
  __syncthreads();
  for (int i = threadIdx.y; i < 32; i += blockDim.y)
    dst[(size_t)(c0 + i) * 1024 + r0 + threadIdx.x] = (bf16_t)tile[threadIdx.x][i];
}

// ---------------------------------------------------------------------------
// Kernel 3: fused QKV projection, register-blocked 16x64 per wave.
// M=B*S=4096, N=H*D=1024 (64-wide n-block == one head), K=E=1024.
// mat 0 -> Q (b,h,s,d); mat 1 -> K (b,h,s,d); mat 2 -> V transposed (b,h,d,s).
__global__ void __launch_bounds__(256) qkv_gemm_kernel(bf16_t* __restrict__ ws) {
  const int mtiles = (CB * CS) / 16;  // 256
  const int nblk   = CN / 64;         // 16
  const int per_mat = mtiles * nblk;  // 4096
  int tile = blockIdx.x * (blockDim.x >> 5) + (threadIdx.x >> 5);
  const int mat = tile / per_mat;
  const int rem = tile % per_mat;
  const int m0 = (rem / nblk) * 16;
  const int nb0 = (rem % nblk) * 64;

  const bf16_t* xh = ws + OFF_XH;
  const bf16_t* wT = ws + (mat == 0 ? OFF_WQ : (mat == 1 ? OFF_WK : OFF_WV));

  v8f c[4] = {v8f{}, v8f{}, v8f{}, v8f{}};
#pragma unroll 2
  for (int e0 = 0; e0 < CE; e0 += 32) {
    v16bf a = load_a_bf16(xh + (size_t)m0 * CE + e0, CE);
#pragma unroll
    for (int i = 0; i < 4; ++i) {
      v16bf b = load_bt_bf16(wT + (size_t)(nb0 + 16 * i) * CE + e0, CE);
      c[i] = wmma_bf16(a, b, c[i]);
    }
  }

  const int l = lane_id();
  const int nn = l & 15, rbase = (l >> 4) * 8;
  const int bb = m0 / CS;   // whole tile in one batch (S % 16 == 0)
  const int h = nb0 >> 6;   // 64-wide n-block == one head
  bf16_t* outp = ws + (mat == 0 ? OFF_Q : (mat == 1 ? OFF_K : OFF_VT));
#pragma unroll
  for (int i = 0; i < 4; ++i) {
    const int d = 16 * i + nn;
#pragma unroll
    for (int r = 0; r < 8; ++r) {
      const int s = (m0 % CS) + rbase + r;
      if (mat < 2)
        outp[(((size_t)bb * CH + h) * CS + s) * CD + d] = (bf16_t)c[i][r];
      else
        outp[(((size_t)bb * CH + h) * CD + d) * CS + s] = (bf16_t)c[i][r];
    }
  }
}

// ---------------------------------------------------------------------------
// Kernel 4 (fused): one block per (b,h,q-tile).
//   Phase 1 (8 waves): exp(mask(Q K^T / 8)) tiles for k < q0+16 -> LDS.
//   Phase 2: per-row sums -> inv[r] = 1/(sum+1e-9).
//   Phase 3: waves 0-3 run AV WMMA from LDS (scale by inv folded into C);
//            waves 4-7 stream normalized weights (NT stores) to global.
// Causal mask is triu(diag=0): j >= i masked (token cannot attend to itself).
constexpr int LDSP = 2052;  // padded row stride in floats (breaks 64-bank conflicts)

__global__ void __launch_bounds__(256)
attn_fused_kernel(bf16_t* __restrict__ ws, const unsigned char* __restrict__ pad,
                  float* __restrict__ wout) {
  __shared__ float lds_ex[16 * LDSP];  // ~128 KB exp-score strip
  __shared__ float red[256];
  __shared__ float inv[16];

  const int qt_count = CS / 16;  // 128
  const int bh = blockIdx.x / qt_count;
  const int q0 = (blockIdx.x % qt_count) * 16;
  const int b = bh / CH;
  const int hh = bh % CH;
  const int wave = threadIdx.x >> 5;
  const int l = lane_id();
  const int nn = l & 15, rbase = (l >> 4) * 8;

  const int kcomp = (q0 + 16 < CS) ? (q0 + 16) : CS;  // active columns (mult of 16)
  const int nkt = kcomp >> 4;                         // active 16-col tiles

  // Zero the 16-col LDS remainder so the 32-wide AV K-loop reads zeros.
  if ((kcomp & 31) && kcomp < CS) {
    const int rr = threadIdx.x >> 4, cc = threadIdx.x & 15;
    lds_ex[rr * LDSP + kcomp + cc] = 0.0f;
  }

  // ---- Phase 1: exp-score tiles into LDS --------------------------------
  const bf16_t* Q = ws + OFF_Q + ((size_t)bh * CS + q0) * CD;
  bool padq[8];
#pragma unroll
  for (int r = 0; r < 8; ++r) padq[r] = pad[b * CS + q0 + rbase + r] != 0;

  for (int kt = wave; kt < nkt; kt += 8) {
    const int k0 = kt * 16;
    const bf16_t* Kp = ws + OFF_K + ((size_t)bh * CS + k0) * CD;
    v8f c = {};
#pragma unroll
    for (int d0 = 0; d0 < CD; d0 += 32) {
      v16bf a = load_a_bf16(Q + d0, CD);
      v16bf bmat = load_bt_bf16(Kp + d0, CD);
      c = wmma_bf16(a, bmat, c);
    }
    const int kj = k0 + nn;
    const bool padk = pad[b * CS + kj] != 0;
#pragma unroll
    for (int r = 0; r < 8; ++r) {
      const int qi = q0 + rbase + r;
      const bool masked = (kj >= qi) || padk || padq[r];
      lds_ex[(rbase + r) * LDSP + kj] = masked ? 0.0f : __expf(c[r] * 0.125f);
    }
  }
  __syncthreads();

  // ---- Phase 2: row sums -> inv -----------------------------------------
  {
    const int rr = threadIdx.x >> 4, cc = threadIdx.x & 15;
    float s = 0.0f;
    for (int col = cc; col < kcomp; col += 16) s += lds_ex[rr * LDSP + col];
    red[threadIdx.x] = s;
    __syncthreads();
    if (cc < 8) red[threadIdx.x] += red[threadIdx.x + 8];
    __syncthreads();
    if (cc < 4) red[threadIdx.x] += red[threadIdx.x + 4];
    __syncthreads();
    if (cc == 0)
      inv[rr] = 1.0f / (red[threadIdx.x] + red[threadIdx.x + 1] +
                        red[threadIdx.x + 2] + red[threadIdx.x + 3] + 1e-9f);
    __syncthreads();
  }

  // ---- Phase 3 ----------------------------------------------------------
  if (wave < 4) {
    // AV GEMM from LDS (unnormalized exp); fold inv into the accumulator.
    const int n0 = wave * 16;
    const bf16_t* Vt = ws + OFF_VT + (size_t)bh * CD * CS;
    v8f c = {};
    for (int k0 = 0; k0 < kcomp; k0 += 32) {
      v16bf a;
      const float* row = &lds_ex[(l & 15) * LDSP + k0 + 8 * (l >> 4)];
#pragma unroll
      for (int j = 0; j < 8; ++j) {
        a[j]     = (bf16_t)row[j];
        a[j + 8] = (bf16_t)row[16 + j];
      }
      v16bf bmat = load_bt_bf16(Vt + (size_t)n0 * CS + k0, CS);
      c = wmma_bf16(a, bmat, c);
    }
    bf16_t* av = ws + OFF_AV;
#pragma unroll
    for (int r = 0; r < 8; ++r) {
      const int m = rbase + r;
      const int s = q0 + m;
      av[((size_t)b * CS + s) * CN + hh * CD + n0 + nn] = (bf16_t)(c[r] * inv[m]);
    }
  } else {
    // Stream normalized weights (active region) and zeros (masked tail).
    // Write-once data: non-temporal stores keep 537 MB out of L2's footprint.
    float* wp = wout + ((size_t)bh * CS + q0) * CS;
    const int t = threadIdx.x - 128;  // 0..127
    for (int idx = t; idx < 16 * CS; idx += 128) {
      const int rr = idx >> 11;        // / 2048
      const int col = idx & (CS - 1);
      const float v = (col < kcomp) ? lds_ex[rr * LDSP + col] * inv[rr] : 0.0f;
      __builtin_nontemporal_store(v, &wp[(size_t)rr * CS + col]);
    }
  }
}

// ---------------------------------------------------------------------------
// Kernel 5: out = AV @ w_o, register-blocked 16x64 per wave.
// M=B*S=4096, N=E=1024, K=H*D=1024.
__global__ void __launch_bounds__(256)
out_gemm_kernel(const bf16_t* __restrict__ ws, float* __restrict__ out) {
  const int nblk = CE / 64;  // 16
  int tile = blockIdx.x * (blockDim.x >> 5) + (threadIdx.x >> 5);
  const int m0 = (tile / nblk) * 16;
  const int nb0 = (tile % nblk) * 64;

  const bf16_t* A = ws + OFF_AV;
  const bf16_t* WoT = ws + OFF_WO;

  v8f c[4] = {v8f{}, v8f{}, v8f{}, v8f{}};
#pragma unroll 2
  for (int k0 = 0; k0 < CN; k0 += 32) {
    v16bf a = load_a_bf16(A + (size_t)m0 * CN + k0, CN);
#pragma unroll
    for (int i = 0; i < 4; ++i) {
      v16bf bmat = load_bt_bf16(WoT + (size_t)(nb0 + 16 * i) * CN + k0, CN);
      c[i] = wmma_bf16(a, bmat, c[i]);
    }
  }

  const int l = lane_id();
  const int nn = l & 15, rbase = (l >> 4) * 8;
#pragma unroll
  for (int i = 0; i < 4; ++i) {
#pragma unroll
    for (int r = 0; r < 8; ++r) {
      const int m = m0 + rbase + r;
      __builtin_nontemporal_store(c[i][r], &out[(size_t)m * CE + nb0 + 16 * i + nn]);
    }
  }
}

// ---------------------------------------------------------------------------
extern "C" void kernel_launch(void* const* d_in, const int* in_sizes, int n_in,
                              void* d_out, int out_size, void* d_ws, size_t ws_size,
                              hipStream_t stream) {
  const float* x = (const float*)d_in[0];
  const unsigned char* pad = (const unsigned char*)d_in[1];  // jnp bool -> 1 byte
  const float* wq = (const float*)d_in[2];
  const float* wk = (const float*)d_in[3];
  const float* wv = (const float*)d_in[4];
  const float* wo = (const float*)d_in[5];

  float* out = (float*)d_out;
  float* weights = out + OUT_ELEMS;
  bf16_t* ws = (bf16_t*)d_ws;  // needs ~48 MB of scratch

  // 1) x -> bf16
  {
    const int n = CB * CS * CE;
    cvt_x_kernel<<<(n + 255) / 256, 256, 0, stream>>>(x, ws + OFF_XH, n);
  }
  // 2) weight matrices transpose-convert -> bf16
  cvt_wT_kernel<<<dim3(32, 32, 4), dim3(32, 8), 0, stream>>>(wq, wk, wv, wo, ws);
  // 3) QKV projections (16x64 per wave)
  {
    const int tiles = 3 * ((CB * CS) / 16) * (CN / 64);  // 12288
    qkv_gemm_kernel<<<tiles / 8, 256, 0, stream>>>(ws);
  }
  // 4) fused scores -> softmax -> AV (+ weights output, single pass)
  {
    const int blocks = CB * CH * (CS / 16);  // 4096
    attn_fused_kernel<<<blocks, 256, 0, stream>>>(ws, pad, weights);
  }
  // 5) output projection (16x64 per wave)
  {
    const int tiles = ((CB * CS) / 16) * (CE / 64);  // 4096
    out_gemm_kernel<<<tiles / 8, 256, 0, stream>>>(ws, out);
  }
}
